// ScaledDotProductAttention_15908558865255
// MI455X (gfx1250) — matrix operand order repaired and checked by reading, not compiled, not run on hardware
//
#include <hip/hip_runtime.h>

// CDNA5 / gfx1250 scaled-dot-product attention (no scale factor, per reference).
//   context  = softmax(Q K^T) V        [16,4096,64]   -> d_out[0 ..)
//   attention= softmax(Q K^T)          [16,4096,4096] -> d_out[16*4096*64 ..)
// Memory-bound on the 1.07 GB attention write (~46us floor @ 23.3 TB/s).
// Matrix math via v_wmma_f32_16x16x32_bf16 (f32 accumulate).

typedef __attribute__((ext_vector_type(16))) __bf16 bf16x16;
typedef __attribute__((ext_vector_type(8)))  float  floatx8;
typedef __attribute__((ext_vector_type(4)))  float  f32x4;

#define B_SZ 16
#define N_SZ 4096
#define D_SZ 64
#define MT   16          // query rows per workgroup
#define NT   (N_SZ / 16) // 256 key tiles
#define WAVES 8
#define ROWSTRIDE 4100   // floats; pad so row m and m+8 hit different LDS banks

extern "C" __global__ __launch_bounds__(256, 1)
void sdpa_fwd_gfx1250(const float* __restrict__ q,
                      const float* __restrict__ k,
                      const float* __restrict__ v,
                      float* __restrict__ ctx,
                      float* __restrict__ att)
{
    extern __shared__ char smem_raw[];
    float* sS   = (float*)smem_raw;                               // [16][ROWSTRIDE] scores -> exp values
    float* sC   = (float*)(smem_raw + (size_t)MT * ROWSTRIDE * 4);// [4][32][8] PV k-half partials
    float* sInv = sC + 4 * 32 * 8;                                // [16] 1/rowsum

    const int tid  = threadIdx.x;
    const int wave = tid >> 5;
    const int lane = tid & 31;
    const int g    = lane >> 4;   // half-wave group (0/1)
    const int h    = lane & 15;

    const int bidx  = blockIdx.x >> 8;    // batch
    const int mtile = blockIdx.x & 255;   // 16-row query tile
    const int m0    = mtile * MT;

    const float* Q = q + ((size_t)bidx * N_SZ + m0) * D_SZ;
    const float* K = k + (size_t)bidx * N_SZ * D_SZ;
    const float* V = v + (size_t)bidx * N_SZ * D_SZ;

    // ---------------- A fragments: Q tile, row m = h ----------------
    // A k order chosen here (not the ISA table order): per-lane k runs [16g, 16g+16) (contiguous, matches B).
    bf16x16 aq[2];
    {
        const float* qrow = Q + (size_t)h * D_SZ;
        #pragma unroll
        for (int c = 0; c < 2; ++c) {
            const f32x4* p0 = (const f32x4*)(qrow + c * 32 + g * 16);
            const f32x4* p1 = (const f32x4*)(qrow + c * 32 + g * 16 + 8);
            f32x4 f0 = p0[0], f1 = p0[1], f2 = p1[0], f3 = p1[1];
            #pragma unroll
            for (int e = 0; e < 4; ++e) {
                aq[c][e]      = (__bf16)f0[e];
                aq[c][4 + e]  = (__bf16)f1[e];
                aq[c][8 + e]  = (__bf16)f2[e];
                aq[c][12 + e] = (__bf16)f3[e];
            }
        }
    }

    // ---------------- Phase 1: S = Q K^T  (f32 scores into LDS) ----------------
    // B k order chosen here (same as A, not the ISA table order): lane n = h; lanes 0-15 hold K=0..15, lanes 16-31 hold K=16..31.
    // B(k,n) = K[n][k]  => per lane a contiguous 16-float run of K's row n.
    for (int t = wave; t < NT; t += WAVES) {
        const float* krow = K + (size_t)(t * 16 + h) * D_SZ;
        bf16x16 b0, b1;
        {
            const f32x4* kp = (const f32x4*)(krow + g * 16);
            f32x4 f0 = kp[0], f1 = kp[1], f2 = kp[2], f3 = kp[3];
            #pragma unroll
            for (int e = 0; e < 4; ++e) {
                b0[e]      = (__bf16)f0[e];
                b0[4 + e]  = (__bf16)f1[e];
                b0[8 + e]  = (__bf16)f2[e];
                b0[12 + e] = (__bf16)f3[e];
            }
        }
        {
            const f32x4* kp = (const f32x4*)(krow + 32 + g * 16);
            f32x4 f0 = kp[0], f1 = kp[1], f2 = kp[2], f3 = kp[3];
            #pragma unroll
            for (int e = 0; e < 4; ++e) {
                b1[e]      = (__bf16)f0[e];
                b1[4 + e]  = (__bf16)f1[e];
                b1[8 + e]  = (__bf16)f2[e];
                b1[12 + e] = (__bf16)f3[e];
            }
        }
        floatx8 c = {};
        c = __builtin_amdgcn_wmma_f32_16x16x32_bf16(false, aq[0], false, b0, (short)0, c, false, false);
        c = __builtin_amdgcn_wmma_f32_16x16x32_bf16(false, aq[1], false, b1, (short)0, c, false, false);
        // C layout: VGPR r -> (m = r + 8g, n = h)
        #pragma unroll
        for (int r = 0; r < 8; ++r)
            sS[(size_t)(r + 8 * g) * ROWSTRIDE + t * 16 + h] = c[r];
    }
    __syncthreads();

    // ---------------- Phase 2: exact softmax; stream attention out ----------------
    // Wave w owns rows 2w, 2w+1.
    #pragma unroll
    for (int rr = 0; rr < 2; ++rr) {
        const int m = wave * 2 + rr;
        float* row = sS + (size_t)m * ROWSTRIDE;

        float mx = -3.0e38f;
        for (int c = lane; c < N_SZ; c += 32) mx = fmaxf(mx, row[c]);
        #pragma unroll
        for (int off = 16; off > 0; off >>= 1) mx = fmaxf(mx, __shfl_xor(mx, off, 32));

        float sum = 0.f;
        for (int c = lane; c < N_SZ; c += 32) {
            float e = __expf(row[c] - mx);
            row[c] = e;                 // keep unnormalized e for PV
            sum += e;
        }
        #pragma unroll
        for (int off = 16; off > 0; off >>= 1) sum += __shfl_xor(sum, off, 32);

        const float inv = __builtin_amdgcn_rcpf(sum);
        if (lane == 0) sInv[m] = inv;

        // normalized attention row -> HBM, nontemporal 16B stores (pure streaming)
        f32x4* arow = (f32x4*)(att + ((size_t)bidx * N_SZ + (m0 + m)) * (size_t)N_SZ);
        const f32x4* rsrc = (const f32x4*)row;
        for (int c4 = lane; c4 < N_SZ / 4; c4 += 32) {
            f32x4 e4 = rsrc[c4] * inv;
            __builtin_nontemporal_store(e4, arow + c4);
        }
    }
    __syncthreads();

    // ---------------- Phase 3: context = P V (P = e-values in LDS) ----------------
    // wave -> (d-tile = w&3, k-half = w>>2); reduce halves through LDS.
    {
        const int dt = wave & 3;
        const int kh = wave >> 2;
        const int d0 = dt * 16;

        floatx8 c = {};
        const int kbeg = kh * (N_SZ / 2);
        for (int kb = kbeg; kb < kbeg + N_SZ / 2; kb += 32) {
            // A fragment: P row m = h, k runs [kb+16g, +16)
            bf16x16 ap;
            const float* prow = sS + (size_t)h * ROWSTRIDE + kb + 16 * g;
            #pragma unroll
            for (int e = 0; e < 8; ++e) ap[e]     = (__bf16)prow[e];
            #pragma unroll
            for (int e = 0; e < 8; ++e) ap[8 + e] = (__bf16)prow[8 + e];
            // B fragment: V(k, d0+h), k = kb + 16g + e  (stride D_SZ in memory)
            bf16x16 bv;
            const float* vcol = V + (size_t)(kb + 16 * g) * D_SZ + d0 + h;
            #pragma unroll
            for (int e = 0; e < 16; ++e) bv[e] = (__bf16)vcol[(size_t)e * D_SZ];
            c = __builtin_amdgcn_wmma_f32_16x16x32_bf16(false, ap, false, bv, (short)0, c, false, false);
        }

        if (kh == 1) {
            #pragma unroll
            for (int r = 0; r < 8; ++r) sC[(size_t)(dt * 32 + lane) * 8 + r] = c[r];
        }
        __syncthreads();
        if (kh == 0) {
            #pragma unroll
            for (int r = 0; r < 8; ++r) c[r] += sC[(size_t)(dt * 32 + lane) * 8 + r];
            #pragma unroll
            for (int r = 0; r < 8; ++r) {
                const int m = r + 8 * g;
                ctx[((size_t)bidx * N_SZ + m0 + m) * D_SZ + d0 + h] = c[r] * sInv[m];
            }
        }
    }
}

extern "C" void kernel_launch(void* const* d_in, const int* in_sizes, int n_in,
                              void* d_out, int out_size, void* d_ws, size_t ws_size,
                              hipStream_t stream) {
    (void)in_sizes; (void)n_in; (void)out_size; (void)d_ws; (void)ws_size;
    const float* q = (const float*)d_in[0];
    const float* k = (const float*)d_in[1];
    const float* v = (const float*)d_in[2];
    float* ctx = (float*)d_out;
    float* att = (float*)d_out + (size_t)B_SZ * N_SZ * D_SZ;

    const size_t lds_bytes = (size_t)MT * ROWSTRIDE * sizeof(float)  // score rows
                           + 4 * 32 * 8 * sizeof(float)              // PV partials
                           + 16 * sizeof(float);                     // 1/rowsum
    // ~266.6 KB dynamic LDS (<= 320 KB/WGP on CDNA5); opt in explicitly.
    (void)hipFuncSetAttribute((const void*)sdpa_fwd_gfx1250,
                              hipFuncAttributeMaxDynamicSharedMemorySize,
                              (int)lds_bytes);

    dim3 grid(B_SZ * (N_SZ / MT));  // 4096 workgroups
    dim3 block(32 * WAVES);         // 8 waves (wave32)
    sdpa_fwd_gfx1250<<<grid, block, lds_bytes, stream>>>(q, k, v, ctx, att);
}
